// StockTransformerV3_38946763440227
// MI455X (gfx1250) — compile-verified, matrix operand order checked by reading, not run on hardware
//
#include <hip/hip_runtime.h>
#include <math.h>

// Problem constants (from reference): B=8, T=512, NF=64, D=1024, H=16, L=6, DF=4096, DH=64
#define BB 8
#define TT 512
#define NF 64
#define DD 1024
#define HH 16
#define LL 6
#define DFF 4096
#define MM (BB * TT) // 4096 rows of activations

typedef __bf16 bfloat;
typedef __attribute__((ext_vector_type(16))) __bf16 v16bf;
typedef __attribute__((ext_vector_type(8)))  __bf16 v8bf;
typedef __attribute__((ext_vector_type(8)))  float  v8f;

__device__ __forceinline__ v8f zero8() {
  v8f z;
#pragma unroll
  for (int i = 0; i < 8; ++i) z[i] = 0.0f;
  return z;
}

// Load a 16-bit A/B WMMA fragment from LDS: two contiguous 8-element (16B) chunks
// at p[0..7] and p[16..23]. Caller passes p = &row[half*8] so that
// half=0 covers K 0-7 / 16-23 and half=1 covers K 8-15 / 24-31 (ISA §7.12.2).
__device__ __forceinline__ v16bf ld_frag_lds(const bfloat* p) {
  union { v16bf v; v8bf h[2]; } u;
  u.h[0] = *(const v8bf*)(p);
  u.h[1] = *(const v8bf*)(p + 16);
  return u.v;
}

// Build an A fragment directly from two global f32 chunks (used for Q), with scale.
__device__ __forceinline__ v16bf cvt_frag(const float* s0, const float* s1, float scale) {
  v16bf r;
#pragma unroll
  for (int i = 0; i < 8; ++i) {
    r[i]     = (bfloat)(s0[i] * scale);
    r[i + 8] = (bfloat)(s1[i] * scale);
  }
  return r;
}

__device__ __forceinline__ float gelu_exact(float x) {
  return 0.5f * x * (1.0f + erff(x * 0.70710678118654752f));
}

__device__ __forceinline__ v8f wmma_bf16(v16bf a, v16bf b, v8f c) {
  // D = A(16x32 bf16) x B(32x16 bf16) + C(16x16 f32)
  return __builtin_amdgcn_wmma_f32_16x16x32_bf16(false, a, false, b, (short)0, c,
                                                 false, false);
}

// ---------------------------------------------------------------------------
// Generic bf16-WMMA GEMM: C[M,N] = epi(A[M,K] @ Bw[K,N] + bias [+ extra])
// Block tile 128x128, 256 threads (8 waves, each owning 32x64 = 2x4 WMMA C frags).
// Double-buffered LDS + software-pipelined global loads: one barrier per K-step;
// next tile's VMEM loads overlap current tile's WMMAs.
// ---------------------------------------------------------------------------
enum { EPI_BIAS = 0, EPI_PE = 1, EPI_GELU = 2, EPI_RELU = 3, EPI_RESID = 4 };

template <int EPI>
__global__ __launch_bounds__(256) void gemm_wmma_kernel(
    const float* __restrict__ A, const float* __restrict__ Bw,
    const float* __restrict__ bias, const float* __restrict__ extra,
    float* __restrict__ C, int Mdim, int Ndim, int Kdim) {
  __shared__ __align__(16) bfloat As[2][128][40];  // [buf][m][k], 80B row stride
  __shared__ __align__(16) bfloat Bs[2][128][40];  // [buf][n][k] (B transposed)

  const int tid  = threadIdx.x;
  const int lane = tid & 31, wave = tid >> 5;
  const int lr   = lane & 15, half = lane >> 4;
  const int wm   = (wave >> 1) * 32;   // 0,32,64,96
  const int wn   = (wave & 1) * 64;    // 0,64
  const int m0   = blockIdx.y * 128, n0 = blockIdx.x * 128;

  v8f acc[2][4];
#pragma unroll
  for (int mi = 0; mi < 2; ++mi)
#pragma unroll
    for (int ni = 0; ni < 4; ++ni) acc[mi][ni] = zero8();

  float4 ra[4], rb[4];  // staged next tile (A: 128x32, B: 32x128, 16 floats each)

  auto load_regs = [&](int k0) {
#pragma unroll
    for (int i = 0; i < 4; ++i) {
      int l = tid + i * 256;
      int row = l >> 3, c4 = (l & 7) << 2;
      int gm = m0 + row;
      float4 f = make_float4(0.f, 0.f, 0.f, 0.f);
      if (gm < Mdim) f = *(const float4*)(A + (size_t)gm * Kdim + k0 + c4);
      ra[i] = f;
    }
#pragma unroll
    for (int i = 0; i < 4; ++i) {
      int l = tid + i * 256;
      int kk = l >> 5, c4 = (l & 31) << 2;
      int gn = n0 + c4;
      const float* src = Bw + (size_t)(k0 + kk) * Ndim + gn;
      float4 f;
      if (gn + 3 < Ndim) {
        f = *(const float4*)src;
      } else {
        f.x = (gn + 0 < Ndim) ? src[0] : 0.f;
        f.y = (gn + 1 < Ndim) ? src[1] : 0.f;
        f.z = (gn + 2 < Ndim) ? src[2] : 0.f;
        f.w = (gn + 3 < Ndim) ? src[3] : 0.f;
      }
      rb[i] = f;
    }
  };

  auto store_lds = [&](int buf) {
#pragma unroll
    for (int i = 0; i < 4; ++i) {
      int l = tid + i * 256;
      int row = l >> 3, c4 = (l & 7) << 2;
      union { bfloat b[4]; uint2 u; } pk;
      pk.b[0] = (bfloat)ra[i].x; pk.b[1] = (bfloat)ra[i].y;
      pk.b[2] = (bfloat)ra[i].z; pk.b[3] = (bfloat)ra[i].w;
      *(uint2*)&As[buf][row][c4] = pk.u;  // 8B-aligned packed store
    }
#pragma unroll
    for (int i = 0; i < 4; ++i) {
      int l = tid + i * 256;
      int kk = l >> 5, c4 = (l & 31) << 2;
      Bs[buf][c4 + 0][kk] = (bfloat)rb[i].x;  // transposed: strided stores
      Bs[buf][c4 + 1][kk] = (bfloat)rb[i].y;
      Bs[buf][c4 + 2][kk] = (bfloat)rb[i].z;
      Bs[buf][c4 + 3][kk] = (bfloat)rb[i].w;
    }
  };

  const int nk = Kdim >> 5;  // K is always a multiple of 32 here
  load_regs(0);
  store_lds(0);
  for (int ki = 0; ki < nk; ++ki) {
    const int cur = ki & 1;
    if (ki + 1 < nk) load_regs((ki + 1) << 5);  // overlap VMEM with WMMA below
    __syncthreads();                            // buf[cur] complete block-wide
    v16bf af[2], bf[4];
#pragma unroll
    for (int mi = 0; mi < 2; ++mi)
      af[mi] = ld_frag_lds(&As[cur][wm + mi * 16 + lr][half * 8]);
#pragma unroll
    for (int ni = 0; ni < 4; ++ni)
      bf[ni] = ld_frag_lds(&Bs[cur][wn + ni * 16 + lr][half * 8]);
#pragma unroll
    for (int mi = 0; mi < 2; ++mi)
#pragma unroll
      for (int ni = 0; ni < 4; ++ni)
        acc[mi][ni] = wmma_bf16(af[mi], bf[ni], acc[mi][ni]);
    if (ki + 1 < nk) store_lds(cur ^ 1);        // safe: peers past this barrier
  }

  // ---- epilogue: C-layout lane holds col=lr, rows r + 8*half ----
#pragma unroll
  for (int mi = 0; mi < 2; ++mi) {
#pragma unroll
    for (int ni = 0; ni < 4; ++ni) {
      int gn = n0 + wn + ni * 16 + lr;
      if (gn >= Ndim) continue;
      float bv = bias ? bias[gn] : 0.0f;
#pragma unroll
      for (int r = 0; r < 8; ++r) {
        int gm = m0 + wm + mi * 16 + r + half * 8;
        if (gm >= Mdim) continue;
        float vv = acc[mi][ni][r] + bv;
        if (EPI == EPI_PE)    vv += extra[(size_t)(gm & (TT - 1)) * Ndim + gn];
        if (EPI == EPI_GELU)  vv = gelu_exact(vv);
        if (EPI == EPI_RELU)  vv = fmaxf(vv, 0.0f);
        if (EPI == EPI_RESID) vv += extra[(size_t)gm * Ndim + gn];
        C[(size_t)gm * Ndim + gn] = vv;
      }
    }
  }
}

// ---------------------------------------------------------------------------
// Flash-style causal attention with head gating.
// Grid (T/64, H, B); 128 threads = 4 waves; each wave owns 16 query rows.
// q/k/v layout: [B, T, D] with head h in columns h*64 .. h*64+63.
// out[b, t, h*64+dh] = g[b,h] * softmax(q k^T / sqrt(DH)) v
// ---------------------------------------------------------------------------
__global__ __launch_bounds__(128) void attn_kernel(
    const float* __restrict__ q, const float* __restrict__ k,
    const float* __restrict__ v, const float* __restrict__ g,
    float* __restrict__ out) {
  __shared__ __align__(16) bfloat Ks[32][72];      // [key][dh] (row stride 144B)
  __shared__ __align__(16) bfloat Vs[64][40];      // transposed: [dh][key]
  __shared__ __align__(16) bfloat Ps[4][16][32];   // per-wave P tile [row][key]

  const int qbi = blockIdx.x, hh = blockIdx.y, bb = blockIdx.z;
  const int tid = threadIdx.x, lane = tid & 31, wave = tid >> 5;
  const int lr = lane & 15, half = lane >> 4;
  const int qrow0 = qbi * 64 + wave * 16;

  // Q fragments (16 rows x 64 dh -> two K=32 A-fragments), pre-scaled by 1/sqrt(DH)
  v16bf qfr[2];
  {
    const float* qp = q + ((size_t)bb * TT + (qrow0 + lr)) * DD + hh * 64;
#pragma unroll
    for (int f = 0; f < 2; ++f) {
      const float* s0 = qp + f * 32 + half * 8;
      qfr[f] = cvt_frag(s0, s0 + 16, 0.125f);
    }
  }

  v8f oacc[4];
#pragma unroll
  for (int t = 0; t < 4; ++t) oacc[t] = zero8();
  float mrow[8], lrow[8];
#pragma unroll
  for (int r = 0; r < 8; ++r) { mrow[r] = -INFINITY; lrow[r] = 0.0f; }

  const int kend = qbi * 64 + 64;  // causal: later keys never needed by this block
  for (int j0 = 0; j0 < kend; j0 += 32) {
    __syncthreads();
    // cooperative load K tile (32 keys x 64 dh) and V tile (transposed)
#pragma unroll
    for (int i = 0; i < 4; ++i) {
      int l = tid + i * 128;
      int key = l >> 4, c4 = (l & 15) << 2;
      float4 f = *(const float4*)(k + ((size_t)bb * TT + (j0 + key)) * DD + hh * 64 + c4);
      union { bfloat b[4]; uint2 u; } pk;
      pk.b[0] = (bfloat)f.x; pk.b[1] = (bfloat)f.y;
      pk.b[2] = (bfloat)f.z; pk.b[3] = (bfloat)f.w;
      *(uint2*)&Ks[key][c4] = pk.u;
    }
#pragma unroll
    for (int i = 0; i < 4; ++i) {
      int l = tid + i * 128;
      int key = l >> 4, c4 = (l & 15) << 2;
      float4 f = *(const float4*)(v + ((size_t)bb * TT + (j0 + key)) * DD + hh * 64 + c4);
      Vs[c4 + 0][key] = (bfloat)f.x; Vs[c4 + 1][key] = (bfloat)f.y;
      Vs[c4 + 2][key] = (bfloat)f.z; Vs[c4 + 3][key] = (bfloat)f.w;
    }
    __syncthreads();

    // S = Q @ K^T : two 16x16 fragments (key subtiles), each two K=32 WMMAs
    v8f sacc[2];
    sacc[0] = zero8(); sacc[1] = zero8();
#pragma unroll
    for (int s = 0; s < 2; ++s)
#pragma unroll
      for (int dhh = 0; dhh < 2; ++dhh) {
        v16bf bf = ld_frag_lds(&Ks[s * 16 + lr][dhh * 32 + half * 8]);
        sacc[s] = wmma_bf16(qfr[dhh], bf, sacc[s]);
      }

    // causal mask + online softmax (rows r + 8*half per lane; cols = lr)
#pragma unroll
    for (int r = 0; r < 8; ++r) {
      int qr = qrow0 + r + half * 8;
      float s0 = (j0 + lr      <= qr) ? sacc[0][r] : -INFINITY;
      float s1 = (j0 + 16 + lr <= qr) ? sacc[1][r] : -INFINITY;
      float tmax = fmaxf(s0, s1);
#pragma unroll
      for (int off = 1; off < 16; off <<= 1)
        tmax = fmaxf(tmax, __shfl_xor(tmax, off, 32));
      float newm = fmaxf(mrow[r], tmax);
      float corr = __expf(mrow[r] - newm);
      float p0 = __expf(s0 - newm);
      float p1 = __expf(s1 - newm);
      float ps = p0 + p1;
#pragma unroll
      for (int off = 1; off < 16; off <<= 1)
        ps += __shfl_xor(ps, off, 32);
      lrow[r] = lrow[r] * corr + ps;
      mrow[r] = newm;
#pragma unroll
      for (int t = 0; t < 4; ++t) oacc[t][r] *= corr;
      Ps[wave][r + half * 8][lr]      = (bfloat)p0;
      Ps[wave][r + half * 8][16 + lr] = (bfloat)p1;
    }
    __syncthreads();

    // O += P @ V  (four 16x16 dh tiles)
    v16bf pfr = ld_frag_lds(&Ps[wave][lr][half * 8]);
#pragma unroll
    for (int t = 0; t < 4; ++t) {
      v16bf vf = ld_frag_lds(&Vs[t * 16 + lr][half * 8]);
      oacc[t] = wmma_bf16(pfr, vf, oacc[t]);
    }
  }

  // final scale: gate / row-sum, store interleaved by head
  const float gs = g[bb * HH + hh];
#pragma unroll
  for (int r = 0; r < 8; ++r) {
    int qr = qrow0 + r + half * 8;
    float sc = gs / lrow[r];
    float* op = out + ((size_t)bb * TT + qr) * DD + hh * 64;
#pragma unroll
    for (int t = 0; t < 4; ++t) op[t * 16 + lr] = oacc[t][r] * sc;
  }
}

// ---------------------------------------------------------------------------
// LayerNorm over D=1024: one block (256 thr) per row
// ---------------------------------------------------------------------------
__global__ __launch_bounds__(256) void ln_kernel(const float* __restrict__ x,
                                                 const float* __restrict__ gamma,
                                                 const float* __restrict__ beta,
                                                 float* __restrict__ y) {
  __shared__ float2 red[256];
  const int row = blockIdx.x, tid = threadIdx.x;
  const float* xr = x + (size_t)row * DD;
  float vals[4];
  float s = 0.f, ss = 0.f;
#pragma unroll
  for (int j = 0; j < 4; ++j) {
    float v = xr[tid + j * 256];
    vals[j] = v; s += v; ss += v * v;
  }
  red[tid] = make_float2(s, ss);
  __syncthreads();
  for (int st = 128; st > 0; st >>= 1) {
    if (tid < st) { red[tid].x += red[tid + st].x; red[tid].y += red[tid + st].y; }
    __syncthreads();
  }
  float mean = red[0].x * (1.0f / DD);
  float var  = red[0].y * (1.0f / DD) - mean * mean;
  float rstd = rsqrtf(var + 1e-5f);
  float* yr = y + (size_t)row * DD;
#pragma unroll
  for (int j = 0; j < 4; ++j) {
    int c = tid + j * 256;
    yr[c] = (vals[j] - mean) * rstd * gamma[c] + beta[c];
  }
}

// mean over T: out[b*D+d] = mean_t h[b,t,d]   (8192 threads)
__global__ void meant_kernel(const float* __restrict__ h, float* __restrict__ out) {
  int idx = blockIdx.x * blockDim.x + threadIdx.x;
  int b = idx >> 10, d = idx & 1023;
  const float* p = h + (size_t)b * TT * DD + d;
  float s = 0.f;
  for (int t = 0; t < TT; ++t) s += p[(size_t)t * DD];
  out[idx] = s * (1.0f / TT);
}

// softmax over H=16 head-gate logits, one thread per batch row
__global__ void gsoftmax_kernel(const float* __restrict__ glog, float* __restrict__ g) {
  int b = threadIdx.x;
  if (b >= BB) return;
  const float* r = glog + b * HH;
  float mx = -INFINITY;
  for (int i = 0; i < HH; ++i) mx = fmaxf(mx, r[i]);
  float e[HH], s = 0.f;
  for (int i = 0; i < HH; ++i) { e[i] = __expf(r[i] - mx); s += e[i]; }
  float inv = 1.0f / s;
  for (int i = 0; i < HH; ++i) g[b * HH + i] = e[i] * inv;
}

// xx += sigmoid(gate) * val from gv[M, 2D]
__global__ void gate_kernel(const float* __restrict__ gv, float* __restrict__ xx) {
  size_t idx = (size_t)blockIdx.x * blockDim.x + threadIdx.x;
  size_t m = idx >> 10;
  int d = (int)(idx & 1023);
  float gt  = gv[m * (2 * DD) + d];
  float val = gv[m * (2 * DD) + DD + d];
  xx[idx] += val * (1.0f / (1.0f + __expf(-gt)));
}

// feat[b] = concat(xx[b,-1], mean_t xx[b], max_t xx[b])
__global__ void pool_kernel(const float* __restrict__ xx, float* __restrict__ feat) {
  int idx = blockIdx.x * blockDim.x + threadIdx.x;  // 8192
  int b = idx >> 10, d = idx & 1023;
  const float* p = xx + (size_t)b * TT * DD + d;
  float s = 0.f, mx = -INFINITY;
  for (int t = 0; t < TT; ++t) {
    float v = p[(size_t)t * DD];
    s += v; mx = fmaxf(mx, v);
  }
  feat[b * 3 * DD + d]          = p[(size_t)(TT - 1) * DD];
  feat[b * 3 * DD + DD + d]     = s * (1.0f / TT);
  feat[b * 3 * DD + 2 * DD + d] = mx;
}

// out[b] = hh2[b,:] @ hW3 + hb3   (K=512, N=1)
__global__ __launch_bounds__(256) void head3_kernel(const float* __restrict__ hh2,
                                                    const float* __restrict__ hW3,
                                                    const float* __restrict__ hb3,
                                                    float* __restrict__ out) {
  __shared__ float red[256];
  int b = blockIdx.x, tid = threadIdx.x;
  const float* r = hh2 + b * (DD / 2);
  red[tid] = r[tid] * hW3[tid] + r[tid + 256] * hW3[tid + 256];
  __syncthreads();
  for (int st = 128; st > 0; st >>= 1) {
    if (tid < st) red[tid] += red[tid + st];
    __syncthreads();
  }
  if (tid == 0) out[b] = red[0] + hb3[0];
}

// ---------------------------------------------------------------------------
extern "C" void kernel_launch(void* const* d_in, const int* in_sizes, int n_in,
                              void* d_out, int out_size, void* d_ws, size_t ws_size,
                              hipStream_t stream) {
  (void)in_sizes; (void)n_in; (void)out_size; (void)ws_size;
  const float* x     = (const float*)d_in[0];
  const float* pe    = (const float*)d_in[1];
  const float* projW = (const float*)d_in[2];
  const float* projb = (const float*)d_in[3];
  const float* Wq    = (const float*)d_in[4];
  const float* bq    = (const float*)d_in[5];
  const float* Wk    = (const float*)d_in[6];
  const float* bk    = (const float*)d_in[7];
  const float* Wv    = (const float*)d_in[8];
  const float* bv    = (const float*)d_in[9];
  const float* Wo    = (const float*)d_in[10];
  const float* bo    = (const float*)d_in[11];
  const float* gW1   = (const float*)d_in[12];
  const float* gb1   = (const float*)d_in[13];
  const float* gW2   = (const float*)d_in[14];
  const float* gb2   = (const float*)d_in[15];
  const float* ln1w  = (const float*)d_in[16];
  const float* ln1b  = (const float*)d_in[17];
  const float* ffW1  = (const float*)d_in[18];
  const float* ffb1  = (const float*)d_in[19];
  const float* ffW2  = (const float*)d_in[20];
  const float* ffb2  = (const float*)d_in[21];
  const float* ln2w  = (const float*)d_in[22];
  const float* ln2b  = (const float*)d_in[23];
  const float* grW   = (const float*)d_in[24];
  const float* grb   = (const float*)d_in[25];
  const float* grlnw = (const float*)d_in[26];
  const float* grlnb = (const float*)d_in[27];
  const float* hW1   = (const float*)d_in[28];
  const float* hb1   = (const float*)d_in[29];
  const float* hW2   = (const float*)d_in[30];
  const float* hb2   = (const float*)d_in[31];
  const float* hW3   = (const float*)d_in[32];
  const float* hb3   = (const float*)d_in[33];
  float* outp = (float*)d_out;

  // workspace carve-up (floats)
  float* ws = (float*)d_ws;
  size_t off = 0;
  float* xx    = ws + off; off += (size_t)MM * DD;   // residual stream
  float* hbuf  = ws + off; off += (size_t)MM * DD;   // post-LN
  float* qb_   = ws + off; off += (size_t)MM * DD;
  float* kb_   = ws + off; off += (size_t)MM * DD;
  float* vb_   = ws + off; off += (size_t)MM * DD;
  float* ob_   = ws + off; off += (size_t)MM * DD;
  float* ffmid = ws + off; off += (size_t)MM * DFF;  // also reused for gv (M x 2D)
  float* gvb   = ffmid;
  float* hmean = ws + off; off += (size_t)BB * DD;
  float* g1b   = ws + off; off += (size_t)BB * DD;
  float* glog  = ws + off; off += (size_t)BB * HH;
  float* gmat  = ws + off; off += (size_t)BB * HH;
  float* feat  = ws + off; off += (size_t)BB * 3 * DD;
  float* hh1   = ws + off; off += (size_t)BB * DD;
  float* hh2   = ws + off; off += (size_t)BB * (DD / 2);

  auto ggrid = [](int Md, int Nd) { return dim3((Nd + 127) / 128, (Md + 127) / 128); };

  // h0 = x @ proj_W + proj_b + pe
  gemm_wmma_kernel<EPI_PE><<<ggrid(MM, DD), 256, 0, stream>>>(
      x, projW, projb, pe, xx, MM, DD, NF);

  for (int l = 0; l < LL; ++l) {
    const size_t wofs = (size_t)l * DD * DD;
    ln_kernel<<<MM, 256, 0, stream>>>(xx, ln1w + l * DD, ln1b + l * DD, hbuf);
    gemm_wmma_kernel<EPI_BIAS><<<ggrid(MM, DD), 256, 0, stream>>>(
        hbuf, Wq + wofs, bq + l * DD, nullptr, qb_, MM, DD, DD);
    gemm_wmma_kernel<EPI_BIAS><<<ggrid(MM, DD), 256, 0, stream>>>(
        hbuf, Wk + wofs, bk + l * DD, nullptr, kb_, MM, DD, DD);
    gemm_wmma_kernel<EPI_BIAS><<<ggrid(MM, DD), 256, 0, stream>>>(
        hbuf, Wv + wofs, bv + l * DD, nullptr, vb_, MM, DD, DD);
    // head gating: softmax(relu(mean_t(h) @ gW1 + gb1) @ gW2 + gb2)
    meant_kernel<<<(BB * DD) / 256, 256, 0, stream>>>(hbuf, hmean);
    gemm_wmma_kernel<EPI_RELU><<<ggrid(BB, DD), 256, 0, stream>>>(
        hmean, gW1 + wofs, gb1 + l * DD, nullptr, g1b, BB, DD, DD);
    gemm_wmma_kernel<EPI_BIAS><<<ggrid(BB, HH), 256, 0, stream>>>(
        g1b, gW2 + (size_t)l * DD * HH, gb2 + l * HH, nullptr, glog, BB, HH, DD);
    gsoftmax_kernel<<<1, 32, 0, stream>>>(glog, gmat);
    // flash attention with gate
    attn_kernel<<<dim3(TT / 64, HH, BB), 128, 0, stream>>>(qb_, kb_, vb_, gmat, ob_);
    // xx += o @ Wo + bo
    gemm_wmma_kernel<EPI_RESID><<<ggrid(MM, DD), 256, 0, stream>>>(
        ob_, Wo + wofs, bo + l * DD, xx, xx, MM, DD, DD);
    // FFN
    ln_kernel<<<MM, 256, 0, stream>>>(xx, ln2w + l * DD, ln2b + l * DD, hbuf);
    gemm_wmma_kernel<EPI_GELU><<<ggrid(MM, DFF), 256, 0, stream>>>(
        hbuf, ffW1 + (size_t)l * DD * DFF, ffb1 + l * DFF, nullptr, ffmid, MM, DFF, DD);
    gemm_wmma_kernel<EPI_RESID><<<ggrid(MM, DD), 256, 0, stream>>>(
        ffmid, ffW2 + (size_t)l * DFF * DD, ffb2 + l * DD, xx, xx, MM, DD, DFF);
  }

  // gated residual: gv = LN(xx) @ grW + grb; xx += sigmoid(gate) * val
  ln_kernel<<<MM, 256, 0, stream>>>(xx, grlnw, grlnb, hbuf);
  gemm_wmma_kernel<EPI_BIAS><<<ggrid(MM, 2 * DD), 256, 0, stream>>>(
      hbuf, grW, grb, nullptr, gvb, MM, 2 * DD, DD);
  gate_kernel<<<(MM * DD) / 256, 256, 0, stream>>>(gvb, xx);

  // pooled features + head MLP
  pool_kernel<<<(BB * DD) / 256, 256, 0, stream>>>(xx, feat);
  gemm_wmma_kernel<EPI_GELU><<<ggrid(BB, DD), 256, 0, stream>>>(
      feat, hW1, hb1, nullptr, hh1, BB, DD, 3 * DD);
  gemm_wmma_kernel<EPI_GELU><<<ggrid(BB, DD / 2), 256, 0, stream>>>(
      hh1, hW2, hb2, nullptr, hh2, BB, DD / 2, DD);
  head3_kernel<<<BB, 256, 0, stream>>>(hh2, hW3, hb3, outp);
}
